// AttentionBlock3D_472446402629
// MI455X (gfx1250) — matrix operand order repaired and checked by reading, not compiled, not run on hardware
//
#include <hip/hip_runtime.h>

#define NBATCH 4
#define NCH    256
#define NSP    4096      // D*H*W
#define NGRP   8
#define CPG    32        // NCH/NGRP
#define EPSV   1e-5f
#define QSCALE 0.0625f   // 256^-0.5

typedef __attribute__((ext_vector_type(16))) _Float16 v16h;
typedef __attribute__((ext_vector_type(8)))  _Float16 v8h;
typedef __attribute__((ext_vector_type(8)))  float    v8f;
typedef __attribute__((ext_vector_type(4)))  unsigned int u32x4;
typedef __attribute__((ext_vector_type(4)))  float    f32x4;

union AFrag { v16h h; u32x4 u[2]; };

static __device__ __forceinline__ v8f wmma_f16(v16h a, v16h b, v8f c) {
  // D = A(16x32 f16) * B(32x16 f16) + C(16x16 f32)
  return __builtin_amdgcn_wmma_f32_16x16x32_f16(false, a, false, b, (short)0, c,
                                                false, false);
}
static __device__ __forceinline__ v8f zero8() {
  v8f z = {0.f,0.f,0.f,0.f,0.f,0.f,0.f,0.f};
  return z;
}

// ---------------------------------------------------------------------------
// CDNA5 async global->LDS copy (ASYNCcnt-tracked), with portable fallback.
// Builtin signature (from compiler diagnostic): param0 is a typed
// int __attribute__((vector_size(16))) pointer in the global address space.
// ---------------------------------------------------------------------------
#if __has_builtin(__builtin_amdgcn_global_load_async_to_lds_b128)
#define USE_ASYNC_LDS 1
#else
#define USE_ASYNC_LDS 0
#endif

typedef int v4i_gcc __attribute__((vector_size(16)));
typedef v4i_gcc __attribute__((address_space(1))) as1_v4i;
typedef v4i_gcc __attribute__((address_space(3))) as3_v4i;

static __device__ __forceinline__ void copy16_to_lds(void* lds_dst,
                                                     const void* gsrc) {
#if USE_ASYNC_LDS
  // Address-space casts via integer round-trip (clang forbids direct
  // pointer casts between address spaces).  Flat->LDS is a low-32-bit
  // truncation on AMDGPU, matching the ISA aperture rule.
  as1_v4i* g = (as1_v4i*)(unsigned long long)gsrc;
  as3_v4i* l = (as3_v4i*)(unsigned int)(unsigned long long)lds_dst;
  __builtin_amdgcn_global_load_async_to_lds_b128(g, l, 0, 0);
#else
  *(u32x4*)lds_dst = *(const u32x4*)gsrc;
#endif
}

static __device__ __forceinline__ void async_wait_all() {
#if USE_ASYNC_LDS
#if __has_builtin(__builtin_amdgcn_s_wait_asynccnt)
  __builtin_amdgcn_s_wait_asynccnt(0);
#else
  asm volatile("s_wait_asynccnt 0" ::: "memory");
#endif
#endif
}

// ---------------------------------------------------------------------------
// Kernel 1: GroupNorm over (C/G, N) per (batch, group); h -> f16 (B, N, C)
// ---------------------------------------------------------------------------
__global__ __launch_bounds__(256) void gn_kernel(const float* __restrict__ x,
    const float* __restrict__ w, const float* __restrict__ bb,
    _Float16* __restrict__ Hs) {
  const int b = blockIdx.x / NGRP, g = blockIdx.x % NGRP;
  const int t = threadIdx.x;
  const float* xp = x + (size_t)(b * NCH + g * CPG) * NSP;
  const int nelem4 = CPG * NSP / 4;

  float s = 0.f, s2 = 0.f;
  for (int i = t; i < nelem4; i += 256) {
    f32x4 v = ((const f32x4*)xp)[i];
    s  += v.x + v.y + v.z + v.w;
    s2 += v.x*v.x + v.y*v.y + v.z*v.z + v.w*v.w;
  }
  __shared__ float r1[256], r2[256];
  r1[t] = s; r2[t] = s2;
  __syncthreads();
  for (int off = 128; off > 0; off >>= 1) {
    if (t < off) { r1[t] += r1[t + off]; r2[t] += r2[t + off]; }
    __syncthreads();
  }
  const float inv  = 1.f / (float)(CPG * NSP);
  const float mu   = r1[0] * inv;
  const float var  = r2[0] * inv - mu * mu;
  const float rstd = rsqrtf(var + EPSV);

  for (int i = t; i < nelem4; i += 256) {
    const int idx = i * 4;
    const int cl = idx >> 12;            // idx / NSP
    const int n  = idx & (NSP - 1);      // idx % NSP
    const int c  = g * CPG + cl;
    const float sw = w[c] * rstd;
    const float sb = bb[c] - mu * sw;
    f32x4 v = ((const f32x4*)xp)[i];
    _Float16* hp = Hs + ((size_t)b * NSP + n) * NCH + c;
    hp[0]        = (_Float16)(v.x * sw + sb);
    hp[NCH]      = (_Float16)(v.y * sw + sb);
    hp[2 * NCH]  = (_Float16)(v.z * sw + sb);
    hp[3 * NCH]  = (_Float16)(v.w * sw + sb);
  }
}

// ---------------------------------------------------------------------------
// Kernel 2: qkv = qkv_w @ h + qkv_b.  One 16x16 tile per wave, K=256 (8 WMMAs).
// Q -> (B,N,C) f16 *scale, K -> (B,N,C) f16, V -> (B,C,N) f16.
// ---------------------------------------------------------------------------
__global__ __launch_bounds__(256) void qkv_kernel(const float* __restrict__ W,
    const float* __restrict__ bias, const _Float16* __restrict__ Hs,
    _Float16* __restrict__ Qs, _Float16* __restrict__ Ks,
    _Float16* __restrict__ Vs) {
  const int b  = blockIdx.z;
  const int o0 = blockIdx.y * 16;
  const int nt = blockIdx.x * 8 + (threadIdx.x >> 5);
  const int n0 = nt * 16;
  const int lm = threadIdx.x & 15;
  const int hh = (threadIdx.x >> 4) & 1;

  v8f acc = zero8();
  const float*    wrow = W  + (size_t)(o0 + lm) * NCH;
  const _Float16* hrow = Hs + ((size_t)b * NSP + n0 + lm) * NCH;
  #pragma unroll
  for (int kc = 0; kc < 8; ++kc) {
    AFrag a, bf;
    const float* wp = wrow + kc * 32 + hh * 16;
    #pragma unroll
    for (int j = 0; j < 8; ++j) {
      a.h[j]     = (_Float16)wp[j];
      a.h[8 + j] = (_Float16)wp[8 + j];
    }
    const u32x4* hp = (const u32x4*)(hrow + kc * 32 + hh * 16);
    bf.u[0] = hp[0]; bf.u[1] = hp[1];
    acc = wmma_f16(a.h, bf.h, acc);
  }

  const int ob = o0 + hh * 8;
  float bs[8];
  #pragma unroll
  for (int i = 0; i < 8; ++i) bs[i] = bias[ob + i];

  if (o0 < NCH) {
    v8h q;
    #pragma unroll
    for (int i = 0; i < 8; ++i) q[i] = (_Float16)((acc[i] + bs[i]) * QSCALE);
    *(v8h*)(Qs + ((size_t)b * NSP + n0 + lm) * NCH + ob) = q;
  } else if (o0 < 2 * NCH) {
    v8h kk;
    #pragma unroll
    for (int i = 0; i < 8; ++i) kk[i] = (_Float16)(acc[i] + bs[i]);
    *(v8h*)(Ks + ((size_t)b * NSP + n0 + lm) * NCH + (ob - NCH)) = kk;
  } else {
    #pragma unroll
    for (int i = 0; i < 8; ++i)
      Vs[((size_t)b * NCH + (ob - 2 * NCH) + i) * NSP + n0 + lm] =
          (_Float16)(acc[i] + bs[i]);
  }
}

// ---------------------------------------------------------------------------
// Kernel 3: flash attention. 4 waves/block, 16 queries/wave, 32-key blocks,
// double-buffered K/V tiles staged via async global->LDS DMA.
// ---------------------------------------------------------------------------
#define KT_STRIDE 264   // 256 + 8 pad (f16 units) -> conflict-free column reads
#define VT_STRIDE 40    // 32 + 8 pad
#define PT_STRIDE 40
#define NKBLK (NSP / 32)

static __device__ __forceinline__ void stage_block(_Float16* KtS, _Float16* VtS,
    const _Float16* __restrict__ Ks, const _Float16* __restrict__ Vs,
    int b, int k0, int tid) {
  // K block: 32 keys x 256 ch, contiguous rows from (B,N,C) layout
  const u32x4* kg = (const u32x4*)(Ks + ((size_t)b * NSP + k0) * NCH);
  for (int i = tid; i < 32 * 32; i += 128) {
    const int r = i >> 5, c = i & 31;
    copy16_to_lds((u32x4*)(KtS + r * KT_STRIDE) + c, kg + r * 32 + c);
  }
  // V block: 256 ch x 32 keys, contiguous rows from (B,C,N) layout
  for (int i = tid; i < NCH * 4; i += 128) {
    const int c = i >> 2, j = i & 3;
    copy16_to_lds((u32x4*)(VtS + c * VT_STRIDE) + j,
                  (const u32x4*)(Vs + ((size_t)b * NCH + c) * NSP + k0) + j);
  }
}

__global__ __launch_bounds__(128) void attn_kernel(
    const _Float16* __restrict__ Qs, const _Float16* __restrict__ Ks,
    const _Float16* __restrict__ Vs, _Float16* __restrict__ Og) {
  const int b   = blockIdx.y;
  const int wv  = threadIdx.x >> 5;
  const int q0  = blockIdx.x * 64 + wv * 16;
  const int lm  = threadIdx.x & 15;
  const int hh  = (threadIdx.x >> 4) & 1;
  const int tid = threadIdx.x;

  __shared__ _Float16 KtS[2][32 * KT_STRIDE];    // [key][channel]
  __shared__ _Float16 VtS[2][NCH * VT_STRIDE];   // [channel][key]
  __shared__ _Float16 PlS[4][16 * PT_STRIDE];    // per-wave P
  _Float16* Pw = &PlS[wv][0];

  // Resident Q fragments (A of S-GEMM), Qs already pre-scaled by C^-1/2.
  AFrag Qf[8];
  {
    const _Float16* qrow = Qs + ((size_t)b * NSP + q0 + lm) * NCH;
    #pragma unroll
    for (int kc = 0; kc < 8; ++kc) {
      const _Float16* p = qrow + kc * 32 + hh * 16;
      Qf[kc].u[0] = *(const u32x4*)p;
      Qf[kc].u[1] = *(const u32x4*)(p + 8);
    }
  }

  v8f acc[16];
  #pragma unroll
  for (int ct = 0; ct < 16; ++ct) acc[ct] = zero8();
  float mrow[8], lrow[8];
  #pragma unroll
  for (int i = 0; i < 8; ++i) { mrow[i] = -1e30f; lrow[i] = 0.f; }

  // Prologue: stage block 0 into buffer 0.
  stage_block(KtS[0], VtS[0], Ks, Vs, b, 0, tid);
  async_wait_all();
  __syncthreads();

  for (int kb = 0; kb < NKBLK; ++kb) {
    const int cur = kb & 1;
    // Prefetch next block into the other buffer (overlaps with compute).
    if (kb + 1 < NKBLK)
      stage_block(KtS[cur ^ 1], VtS[cur ^ 1], Ks, Vs, b, (kb + 1) * 32, tid);

    const _Float16* Kt = KtS[cur];
    const _Float16* Vt = VtS[cur];

    // S = Q^T K for the two 16-key tiles of this block (16 WMMAs)
    v8f s0 = zero8(), s1 = zero8();
    #pragma unroll
    for (int kc = 0; kc < 8; ++kc) {
      AFrag b0, b1;
      const _Float16* kp0 = Kt + (lm)      * KT_STRIDE + kc * 32 + hh * 16;
      const _Float16* kp1 = Kt + (16 + lm) * KT_STRIDE + kc * 32 + hh * 16;
      b0.u[0] = ((const u32x4*)kp0)[0]; b0.u[1] = ((const u32x4*)kp0)[1];
      b1.u[0] = ((const u32x4*)kp1)[0]; b1.u[1] = ((const u32x4*)kp1)[1];
      s0 = wmma_f16(Qf[kc].h, b0.h, s0);
      s1 = wmma_f16(Qf[kc].h, b1.h, s1);
    }

    // Online softmax: row reductions across the 16 lanes of each half-wave
    float alpha[8];
    #pragma unroll
    for (int i = 0; i < 8; ++i) {
      float cmax = fmaxf(s0[i], s1[i]);
      #pragma unroll
      for (int msk = 1; msk < 16; msk <<= 1)
        cmax = fmaxf(cmax, __shfl_xor(cmax, msk, 32));
      const float mn = fmaxf(mrow[i], cmax);
      alpha[i] = __expf(mrow[i] - mn);
      const float p0 = __expf(s0[i] - mn);
      const float p1 = __expf(s1[i] - mn);
      float rs = p0 + p1;
      #pragma unroll
      for (int msk = 1; msk < 16; msk <<= 1) rs += __shfl_xor(rs, msk, 32);
      lrow[i] = lrow[i] * alpha[i] + rs;
      mrow[i] = mn;
      const int row = i + 8 * hh;
      Pw[row * PT_STRIDE + lm]      = (_Float16)p0;
      Pw[row * PT_STRIDE + 16 + lm] = (_Float16)p1;
    }
    #pragma unroll
    for (int ct = 0; ct < 16; ++ct) {
      #pragma unroll
      for (int i = 0; i < 8; ++i) acc[ct][i] *= alpha[i];
    }

    // O += P * V^T (16 WMMAs; P fragment via wave-private LDS transpose)
    AFrag pf;
    {
      const _Float16* pp = Pw + lm * PT_STRIDE + hh * 16;
      pf.u[0] = *(const u32x4*)pp;
      pf.u[1] = *(const u32x4*)(pp + 8);
    }
    #pragma unroll
    for (int ct = 0; ct < 16; ++ct) {
      AFrag vf;
      const _Float16* vp = Vt + (ct * 16 + lm) * VT_STRIDE + hh * 16;
      vf.u[0] = ((const u32x4*)vp)[0];
      vf.u[1] = ((const u32x4*)vp)[1];
      acc[ct] = wmma_f16(pf.h, vf.h, acc[ct]);
    }

    // Next buffer staged + everyone done reading current buffer.
    async_wait_all();
    __syncthreads();
  }

  // Epilogue: O /= l, store f16 (B, N, C)
  float rinv[8];
  #pragma unroll
  for (int i = 0; i < 8; ++i) rinv[i] = 1.f / lrow[i];
  #pragma unroll
  for (int ct = 0; ct < 16; ++ct) {
    #pragma unroll
    for (int i = 0; i < 8; ++i)
      Og[((size_t)b * NSP + q0 + i + 8 * hh) * NCH + ct * 16 + lm] =
          (_Float16)(acc[ct][i] * rinv[i]);
  }
}

// ---------------------------------------------------------------------------
// Kernel 4: y = x + out_w @ O^T + out_b.  R[n,o] tile per wave (8 WMMAs).
// ---------------------------------------------------------------------------
__global__ __launch_bounds__(256) void proj_kernel(const float* __restrict__ x,
    const float* __restrict__ W, const float* __restrict__ bias,
    const _Float16* __restrict__ Og, float* __restrict__ out) {
  const int b    = blockIdx.y;
  const int tile = blockIdx.x * 8 + (threadIdx.x >> 5);
  const int nt   = tile >> 4, ot = tile & 15;
  const int n0   = nt * 16, o0 = ot * 16;
  const int lm   = threadIdx.x & 15;
  const int hh   = (threadIdx.x >> 4) & 1;

  v8f acc = zero8();
  const _Float16* arow = Og + ((size_t)b * NSP + n0 + lm) * NCH;
  const float*    wrow = W + (size_t)(o0 + lm) * NCH;
  #pragma unroll
  for (int kc = 0; kc < 8; ++kc) {
    AFrag a, bf;
    const _Float16* ap = arow + kc * 32 + hh * 16;
    a.u[0] = *(const u32x4*)ap;
    a.u[1] = *(const u32x4*)(ap + 8);
    const float* wp = wrow + kc * 32 + hh * 16;
    #pragma unroll
    for (int j = 0; j < 16; ++j) bf.h[j] = (_Float16)wp[j];
    acc = wmma_f16(a.h, bf.h, acc);
  }

  const int o = o0 + lm;
  const float bv = bias[o];
  const size_t base = ((size_t)b * NCH + o) * NSP + n0 + 8 * hh;
  f32x4 x0 = *(const f32x4*)(x + base);
  f32x4 x1 = *(const f32x4*)(x + base + 4);
  f32x4 r0, r1;
  r0.x = x0.x + acc[0] + bv; r0.y = x0.y + acc[1] + bv;
  r0.z = x0.z + acc[2] + bv; r0.w = x0.w + acc[3] + bv;
  r1.x = x1.x + acc[4] + bv; r1.y = x1.y + acc[5] + bv;
  r1.z = x1.z + acc[6] + bv; r1.w = x1.w + acc[7] + bv;
  *(f32x4*)(out + base)     = r0;
  *(f32x4*)(out + base + 4) = r1;
}

// ---------------------------------------------------------------------------
extern "C" void kernel_launch(void* const* d_in, const int* in_sizes, int n_in,
                              void* d_out, int out_size, void* d_ws,
                              size_t ws_size, hipStream_t stream) {
  const float* x      = (const float*)d_in[0];
  const float* norm_w = (const float*)d_in[1];
  const float* norm_b = (const float*)d_in[2];
  const float* qkv_w  = (const float*)d_in[3];
  const float* qkv_b  = (const float*)d_in[4];
  const float* out_w  = (const float*)d_in[5];
  const float* out_b  = (const float*)d_in[6];
  float* out = (float*)d_out;

  const size_t per_bytes = (size_t)NBATCH * NSP * NCH * sizeof(_Float16);
  char* ws = (char*)d_ws;
  _Float16* Hs = (_Float16*)(ws);
  _Float16* Qs = (_Float16*)(ws + 1 * per_bytes);
  _Float16* Ks = (_Float16*)(ws + 2 * per_bytes);
  _Float16* Vs = (_Float16*)(ws + 3 * per_bytes);
  _Float16* Og = (_Float16*)(ws + 4 * per_bytes);

  gn_kernel<<<dim3(NBATCH * NGRP), dim3(256), 0, stream>>>(x, norm_w, norm_b, Hs);
  qkv_kernel<<<dim3(NSP / 16 / 8, 3 * NCH / 16, NBATCH), dim3(256), 0, stream>>>(
      qkv_w, qkv_b, Hs, Qs, Ks, Vs);
  attn_kernel<<<dim3(NSP / 64, NBATCH), dim3(128), 0, stream>>>(Qs, Ks, Vs, Og);
  proj_kernel<<<dim3((NSP / 16) * (NCH / 16) / 8, NBATCH), dim3(256), 0, stream>>>(
      x, out_w, out_b, Og, out);
}
